// CFI_10101763080504
// MI455X (gfx1250) — matrix operand order repaired and checked by reading, not compiled, not run on hardware
//
#include <hip/hip_runtime.h>
#include <math.h>

typedef __attribute__((ext_vector_type(16))) _Float16 v16h;
typedef __attribute__((ext_vector_type(8)))  _Float16 v8h;
typedef __attribute__((ext_vector_type(8)))  float    v8f;
typedef unsigned int u32x4 __attribute__((ext_vector_type(4)));
typedef int          i32x4 __attribute__((ext_vector_type(4)));
typedef int          i32x8 __attribute__((ext_vector_type(8)));

#define BATCH   32
#define CIN     256
#define HW      3136     // 56*56
#define INITC   128
#define EXPC    128
#define NEXP    4
#define BNEPS   1e-5f

#define NTILE   64       // N columns per block
#define RAWPITCH 66      // 64 floats + 2 pad DWORDs (TDM pad_interval=64dw, pad_amount=2dw)
#define XSPITCH 264      // 256 halves + 8 pad halves (conflict-free b128 reads)

// ---------------------------------------------------------------------------
// Spatial mean pool: one block per (b,c). src channel index = b*chStride + c.
// ---------------------------------------------------------------------------
__global__ __launch_bounds__(256) void pool_kernel(const float* __restrict__ src,
                                                   float* __restrict__ dst,
                                                   int C, int chStride) {
    __shared__ float red[256];
    int bc = blockIdx.x;
    int b  = bc / C;
    int c  = bc - b * C;
    const float* p = src + ((size_t)b * chStride + c) * HW;
    int t = threadIdx.x;
    float s = 0.f;
    for (int i = t; i < HW; i += 256) s += p[i];
    red[t] = s;
    __syncthreads();
    for (int off = 128; off > 0; off >>= 1) {
        if (t < off) red[t] += red[t + off];
        __syncthreads();
    }
    if (t == 0) dst[bc] = red[0] * (1.0f / (float)HW);
}

// ---------------------------------------------------------------------------
// Routing MLP: r[b][e] = sigmoid(pooled[b]·w[e] + bias[e]).  Block per sample.
// ---------------------------------------------------------------------------
__global__ void route_kernel(const float* __restrict__ pooled,
                             const float* __restrict__ w,
                             const float* __restrict__ bias,
                             float* __restrict__ r, int C) {
    int b = blockIdx.x;
    int e = threadIdx.x;
    if (e < NEXP) {
        const float* pv = pooled + (size_t)b * C;
        const float* wv = w + (size_t)e * C;
        float s = bias[e];
        for (int c = 0; c < C; ++c) s += pv[c] * wv[c];
        r[b * NEXP + e] = 1.0f / (1.0f + __expf(-s));
    }
}

// ---------------------------------------------------------------------------
// Mix experts for the 1x1 conv: K1[b][o][c] (f16), row-major M=o, K=c.
// ---------------------------------------------------------------------------
__global__ __launch_bounds__(256) void buildK1_kernel(const float* __restrict__ r1,
                                                      const float* __restrict__ w1,
                                                      _Float16* __restrict__ K1) {
    int idx = blockIdx.x * 256 + threadIdx.x;      // b*INITC*CIN + o*CIN + c
    int b = idx / (INITC * CIN);
    int oc = idx - b * (INITC * CIN);              // o*CIN + c
    float acc = 0.f;
    #pragma unroll
    for (int e = 0; e < NEXP; ++e)
        acc += r1[b * NEXP + e] * w1[(size_t)e * INITC * CIN + oc];
    K1[idx] = (_Float16)acc;
}

// ---------------------------------------------------------------------------
// Mix experts for the depthwise 3x3: K2[b][c][9] (f32).
// ---------------------------------------------------------------------------
__global__ __launch_bounds__(256) void buildK2_kernel(const float* __restrict__ r2,
                                                      const float* __restrict__ w2,
                                                      float* __restrict__ K2) {
    int idx = blockIdx.x * 256 + threadIdx.x;
    if (idx >= BATCH * EXPC * 9) return;
    int b = idx / (EXPC * 9);
    int cj = idx - b * (EXPC * 9);                 // c*9 + j
    float acc = 0.f;
    #pragma unroll
    for (int e = 0; e < NEXP; ++e)
        acc += r2[b * NEXP + e] * w2[(size_t)e * EXPC * 9 + cj];
    K2[idx] = acc;
}

// ---------------------------------------------------------------------------
// Main per-sample GEMM via WMMA: Y1[b] = K1[b](128x256) * X[b](256x3136).
// Block tile: M=128, N=64.  8 waves; wave w owns M-tile [16w,16w+16) and all
// four N-subtiles (32 WMMAs/wave, A fragment reused 4x).  X tile staged into
// LDS by the Tensor Data Mover (raw f32 tile, TDM-padded rows) then converted
// to f16 in the transposed padded B layout.  Epilogue folds BN + ReLU and
// writes x1 to out channels [0,128).
// ---------------------------------------------------------------------------
__global__ __launch_bounds__(256) void gemm1x1_wmma(const float* __restrict__ x,
                                                    const _Float16* __restrict__ K1,
                                                    const float* __restrict__ g1,
                                                    const float* __restrict__ b1,
                                                    const float* __restrict__ m1,
                                                    const float* __restrict__ v1,
                                                    float* __restrict__ out) {
    // raw must be FIRST so its LDS byte offset is 0 (used in the TDM D#).
    __shared__ struct {
        float    raw[CIN * RAWPITCH];              // 256 rows x (64+2 pad) f32
        _Float16 xs[NTILE * XSPITCH];              // [n][k] f16, +8-half row pad
    } S;

    int nt = blockIdx.x;
    int b  = blockIdx.y;
    int n0 = nt * NTILE;
    int t  = threadIdx.x;
    int wave = t >> 5;
    int lane = t & 31;

    const float* gsrc = x + (size_t)b * CIN * HW + n0;   // tile origin [k=0][n=0]

#if __has_builtin(__builtin_amdgcn_tensor_load_to_lds)
    // ---- TDM: DMA the 256x64 f32 tile into LDS with 2-DWORD row padding ----
    if (wave == 0) {
        unsigned long long ga = (unsigned long long)(uintptr_t)gsrc;
        u32x4 g0;
        g0[0] = 1u;                                         // count=1, user mode
        g0[1] = 0u;                                         // lds_addr = S.raw (offset 0)
        g0[2] = (unsigned)(ga & 0xffffffffu);               // global_addr[31:0]
        g0[3] = (unsigned)(ga >> 32) | (2u << 30);          // addr[56:32] | type=2
        i32x8 g1d;
        g1d[0] = (2 << 16)      // data_size = 4B
               | (1 << 20)      // pad_enable
               | (5 << 22)      // pad_interval: every 64 DWORDs
               | (1 << 25);     // pad_amount: 2 DWORDs
        g1d[1] = (HW & 0xffff) << 16;                       // tensor_dim0 = 3136 (lo16)
        g1d[2] = (CIN & 0xffff) << 16;                      // dim0 hi=0 | tensor_dim1 = 256
        g1d[3] = (NTILE << 16);                             // dim1 hi=0 | tile_dim0 = 64
        g1d[4] = CIN;                                       // tile_dim1 = 256, tile_dim2 = 0
        g1d[5] = HW;                                        // tensor_dim0_stride = 3136
        g1d[6] = 0;                                         // stride hi | dim1_stride lo
        g1d[7] = 0;
        i32x4 zz = {0, 0, 0, 0};
#if defined(__clang_major__) && __clang_major__ >= 23
        i32x8 z8 = {0, 0, 0, 0, 0, 0, 0, 0};
        __builtin_amdgcn_tensor_load_to_lds(g0, g1d, zz, zz, z8, 0);
#else
        __builtin_amdgcn_tensor_load_to_lds(g0, g1d, zz, zz, 0);
#endif
        __builtin_amdgcn_s_wait_tensorcnt(0);
    }
    __syncthreads();
    // ---- convert raw f32 tile -> f16 transposed B layout ----
    {
        int k = t;                                          // thread owns K-row k
        const float2* rk = (const float2*)&S.raw[k * RAWPITCH];
        #pragma unroll
        for (int q = 0; q < NTILE / 2; ++q) {
            float2 f = rk[q];
            S.xs[(2 * q + 0) * XSPITCH + k] = (_Float16)f.x;
            S.xs[(2 * q + 1) * XSPITCH + k] = (_Float16)f.y;
        }
    }
#else
    // ---- fallback: stage straight from global ----
    {
        int k = t;
        const float4* p4 = (const float4*)(gsrc + (size_t)k * HW);
        __builtin_prefetch(gsrc + (size_t)k * HW + NTILE, 0, 1);
        #pragma unroll
        for (int q = 0; q < NTILE / 4; ++q) {
            float4 f = p4[q];
            S.xs[(4 * q + 0) * XSPITCH + k] = (_Float16)f.x;
            S.xs[(4 * q + 1) * XSPITCH + k] = (_Float16)f.y;
            S.xs[(4 * q + 2) * XSPITCH + k] = (_Float16)f.z;
            S.xs[(4 * q + 3) * XSPITCH + k] = (_Float16)f.w;
        }
    }
#endif
    __syncthreads();

    int hs  = lane >> 4;                           // half-wave select (0/1)
    int l16 = lane & 15;
    int m0  = wave * 16;

    // A fragment base for this lane: row M = m0+l16, row-major K
    const _Float16* Arow = K1 + ((size_t)b * INITC + m0 + l16) * CIN;

    v8f acc0 = {}, acc1 = {}, acc2 = {}, acc3 = {};
    #pragma unroll
    for (int kb = 0; kb < CIN; kb += 32) {
        // A: 16-bit 16x32 layout — lanes 0-15: K=[kb,kb+8)+[kb+16,kb+24);
        //                          lanes 16-31: K=[kb+8,kb+16)+[kb+24,kb+32)
        union { v16h v; v8h h[2]; } A;
        const v8h* ap = (const v8h*)(Arow + kb + hs * 8);
        A.h[0] = ap[0];
        A.h[1] = ap[2];                            // +16 halves
        // B: lane holds col N = ns*16+l16, K=[kb+8*hs,+8)+[kb+16+8*hs,+8)
        #pragma unroll
        for (int ns = 0; ns < 4; ++ns) {
            union { v16h v; v8h h[2]; } Bf;
            const _Float16* bp = &S.xs[(ns * 16 + l16) * XSPITCH + kb + hs * 8];
            Bf.h[0] = *(const v8h*)bp;
            Bf.h[1] = *(const v8h*)(bp + 16);
            v8f& acc = (ns == 0) ? acc0 : (ns == 1) ? acc1 : (ns == 2) ? acc2 : acc3;
            acc = __builtin_amdgcn_wmma_f32_16x16x32_f16(
                false, A.v, false, Bf.v, (short)0, acc, false, false);
        }
    }

    // ---- epilogue: BN fold + ReLU; D vgpr i -> M = m0+i+8*hs, N col per ns ----
    float inv8[8], beta8[8];
    #pragma unroll
    for (int i = 0; i < 8; ++i) {
        int m = m0 + i + 8 * hs;
        inv8[i]  = g1[m] / sqrtf(v1[m] + BNEPS);
        beta8[i] = b1[m] - m1[m] * inv8[i];
    }
    float* ob = out + (size_t)b * 256 * HW;
    #pragma unroll
    for (int ns = 0; ns < 4; ++ns) {
        const v8f& acc = (ns == 0) ? acc0 : (ns == 1) ? acc1 : (ns == 2) ? acc2 : acc3;
        int ncol = n0 + ns * 16 + l16;
        #pragma unroll
        for (int i = 0; i < 8; ++i) {
            int m = m0 + i + 8 * hs;
            float y = acc[i] * inv8[i] + beta8[i];
            y = y > 0.f ? y : 0.f;
            ob[(size_t)m * HW + ncol] = y;
        }
    }
}

// ---------------------------------------------------------------------------
// Depthwise 3x3 (pad 1) + BN + ReLU. Reads x1 from out[:,0:128), writes
// out[:,128:256). grid = (13 pixel-blocks, 128 channels, 32 samples).
// ---------------------------------------------------------------------------
__global__ __launch_bounds__(256) void dwconv3x3(const float* __restrict__ out_in,
                                                 const float* __restrict__ K2,
                                                 const float* __restrict__ g2,
                                                 const float* __restrict__ b2,
                                                 const float* __restrict__ m2,
                                                 const float* __restrict__ v2,
                                                 float* __restrict__ out) {
    int p = blockIdx.x * 256 + threadIdx.x;
    if (p >= HW) return;
    int c = blockIdx.y;
    int b = blockIdx.z;
    int h = p / 56, w = p - h * 56;
    const float* src = out_in + ((size_t)b * 256 + c) * HW;
    const float* kk  = K2 + ((size_t)b * EXPC + c) * 9;
    float s = 0.f;
    #pragma unroll
    for (int dh = -1; dh <= 1; ++dh) {
        int hh = h + dh;
        if (hh < 0 || hh >= 56) continue;
        #pragma unroll
        for (int dw = -1; dw <= 1; ++dw) {
            int ww = w + dw;
            if (ww < 0 || ww >= 56) continue;
            s += kk[(dh + 1) * 3 + (dw + 1)] * src[hh * 56 + ww];
        }
    }
    float inv  = g2[c] / sqrtf(v2[c] + BNEPS);
    float beta = b2[c] - m2[c] * inv;
    float y = s * inv + beta;
    y = y > 0.f ? y : 0.f;
    out[((size_t)b * 256 + 128 + c) * HW + p] = y;
}

// ---------------------------------------------------------------------------
extern "C" void kernel_launch(void* const* d_in, const int* in_sizes, int n_in,
                              void* d_out, int out_size, void* d_ws, size_t ws_size,
                              hipStream_t stream) {
    (void)in_sizes; (void)n_in; (void)out_size; (void)ws_size;
    const float* x    = (const float*)d_in[0];
    const float* r1_w = (const float*)d_in[1];
    const float* r1_b = (const float*)d_in[2];
    const float* w1   = (const float*)d_in[3];
    const float* g1   = (const float*)d_in[4];
    const float* b1   = (const float*)d_in[5];
    const float* m1   = (const float*)d_in[6];
    const float* v1   = (const float*)d_in[7];
    const float* r2_w = (const float*)d_in[8];
    const float* r2_b = (const float*)d_in[9];
    const float* w2   = (const float*)d_in[10];
    const float* g2   = (const float*)d_in[11];
    const float* b2   = (const float*)d_in[12];
    const float* m2   = (const float*)d_in[13];
    const float* v2   = (const float*)d_in[14];
    float* out = (float*)d_out;

    // workspace carve-up (f32 region then f16 region)
    float* pooled1 = (float*)d_ws;                         // 32*256
    float* r1      = pooled1 + BATCH * CIN;                // 32*4
    float* pooled2 = r1 + BATCH * NEXP;                    // 32*128
    float* r2      = pooled2 + BATCH * INITC;              // 32*4
    float* K2      = r2 + BATCH * NEXP;                    // 32*128*9
    _Float16* K1   = (_Float16*)(K2 + BATCH * EXPC * 9);   // 32*128*256 halves

    // branch 1: pool -> route -> mix experts -> WMMA GEMM (+BN+ReLU)
    pool_kernel<<<dim3(BATCH * CIN), 256, 0, stream>>>(x, pooled1, CIN, 256);
    route_kernel<<<dim3(BATCH), 32, 0, stream>>>(pooled1, r1_w, r1_b, r1, CIN);
    buildK1_kernel<<<dim3(BATCH * INITC * CIN / 256), 256, 0, stream>>>(r1, w1, K1);
    gemm1x1_wmma<<<dim3(HW / NTILE, BATCH), 256, 0, stream>>>(x, K1, g1, b1, m1, v1, out);

    // branch 2: pool(x1) -> route -> mix experts -> depthwise 3x3 (+BN+ReLU)
    pool_kernel<<<dim3(BATCH * INITC), 256, 0, stream>>>(out, pooled2, INITC, 256);
    route_kernel<<<dim3(BATCH), 32, 0, stream>>>(pooled2, r2_w, r2_b, r2, INITC);
    buildK2_kernel<<<dim3((BATCH * EXPC * 9 + 255) / 256), 256, 0, stream>>>(r2, w2, K2);
    dwconv3x3<<<dim3((HW + 255) / 256, EXPC, BATCH), 256, 0, stream>>>(
        out, K2, g2, b2, m2, v2, out);
}
